// RelationalAttention_33011118637835
// MI455X (gfx1250) — matrix operand order repaired and checked by reading, not compiled, not run on hardware
//
#include <hip/hip_runtime.h>
#include <hip/hip_bf16.h>

// ---------------------------------------------------------------------------
// Types for CDNA5 WMMA (wave32): v16bf A/B fragments, v8f C/D accumulators.
// ---------------------------------------------------------------------------
typedef __bf16 bf16;
typedef bf16  bf16x8  __attribute__((ext_vector_type(8)));
typedef bf16  bf16x16 __attribute__((ext_vector_type(16)));
typedef float v8f     __attribute__((ext_vector_type(8)));

// A fragment: 16x32 bf16, element (m,k) at base[m*ld + k].
// Layout (ISA 7.12.2): lanes 0-15 hold row M=lane, K in {0..7, 16..23};
// lanes 16-31 hold row M=lane-16, K in {8..15, 24..31}. ld multiple of 8.
__device__ __forceinline__ bf16x16 frag_a(const bf16* __restrict__ base, int ld) {
    const int lane = threadIdx.x & 31;
    const int r = lane & 15, hi = lane >> 4;
    const bf16* p = base + r * ld + hi * 8;
    bf16x8 lo = *(const bf16x8*)(p);
    bf16x8 hh = *(const bf16x8*)(p + 16);
    bf16x16 f;
#pragma unroll
    for (int i = 0; i < 8; ++i) { f[i] = lo[i]; f[i + 8] = hh[i]; }
    return f;
}

// B fragment: 32x16 bf16, from a transposed tile: element (k,n) at baseT[n*ld + k].
// Lanes 0-15 hold column N=lane, K in {0..7, 16..23}; lanes 16-31 K in {8..15, 24..31}.
__device__ __forceinline__ bf16x16 frag_b(const bf16* __restrict__ baseT, int ld) {
    const int lane = threadIdx.x & 31;
    const int c = lane & 15, hi = lane >> 4;
    const bf16* p = baseT + c * ld + hi * 8;
    bf16x8 lo = *(const bf16x8*)(p);
    bf16x8 hh = *(const bf16x8*)(p + 16);
    bf16x16 f;
#pragma unroll
    for (int i = 0; i < 8; ++i) { f[i] = lo[i]; f[i + 8] = hh[i]; }
    return f;
}

__device__ __forceinline__ v8f wmma_bf16(bf16x16 a, bf16x16 b, v8f c) {
    return __builtin_amdgcn_wmma_f32_16x16x32_bf16(
        /*neg_a=*/false, a, /*neg_b=*/false, b,
        /*c_mod=*/(short)0, c, /*reuse_a=*/false, /*reuse_b=*/false);
}

// ---------------------------------------------------------------------------
// Kernel 0: transpose + convert weights once (L2-resident, tiny).
//   Wqkv (256x768 f32) -> WqkvT (768x256 bf16)
//   Wproj(256x256 f32) -> WprojT(256x256 bf16)
// Grid 1024: one output row (contiguous K) per block.
// ---------------------------------------------------------------------------
__global__ __launch_bounds__(256) void transpose_weights(
    const float* __restrict__ Wqkv, const float* __restrict__ Wproj,
    bf16* __restrict__ WqkvT, bf16* __restrict__ WprojT)
{
    const int row = blockIdx.x;      // output row = original column
    const int kk  = threadIdx.x;     // original row (contraction index)
    if (row < 768) {
        WqkvT[(size_t)row * 256 + kk] = (bf16)Wqkv[(size_t)kk * 768 + row];
    } else {
        const int c = row - 768;
        WprojT[(size_t)c * 256 + kk] = (bf16)Wproj[(size_t)kk * 256 + c];
    }
}

// ---------------------------------------------------------------------------
// Kernel 1: qkv = x(4096x256) @ Wqkv + bqkv   (bf16 WMMA, f32 acc)
// Each block's 64x64 output tile maps to exactly one (b, h) slice of one of
// q(b,h,n,d) / k(b,h,m,d) / v^T(b,h,d,m): epilogue round-trips C through LDS
// and emits fully contiguous bf16x8 global stores (incl. the v transpose).
// Grid (12, 64); 8 waves x 2 WMMA tiles.
// ---------------------------------------------------------------------------
__global__ __launch_bounds__(256) void qkv_gemm(
    const float* __restrict__ x, const bf16* __restrict__ WT,
    const float* __restrict__ bqkv,
    bf16* __restrict__ q, bf16* __restrict__ k, bf16* __restrict__ vt)
{
    __shared__ __align__(16) bf16  sA[64][32];   // x tile (row-major, K contig)
    __shared__ __align__(16) bf16  sBT[64][32];  // W^T tile: [n][k]
    __shared__ __align__(16) float sOut[64][64]; // C round-trip
    __shared__ float sBias[64];
    const int t = threadIdx.x;
    const int wave = t >> 5, lane = t & 31;
    const int row0 = blockIdx.y * 64;
    const int col0 = blockIdx.x * 64;
    const int idx0 = wave * 2, idx1 = idx0 + 1;
    const int tr = idx0 >> 2;                    // tr identical for both tiles
    const int tc0 = idx0 & 3, tc1 = idx1 & 3;
    v8f acc0 = {}, acc1 = {};

    if (t < 64) sBias[t] = bqkv[col0 + t];

    for (int k0 = 0; k0 < 256; k0 += 32) {
        { // stage A (convert f32 -> bf16, one ds_store_b128)
            const int m = t >> 2, kg = (t & 3) * 8;
            const float* src = x + (size_t)(row0 + m) * 256 + k0 + kg;
            bf16x8 tmp;
#pragma unroll
            for (int j = 0; j < 8; ++j) tmp[j] = (bf16)src[j];
            *(bf16x8*)(&sA[m][kg]) = tmp;
        }
        { // stage B^T: already bf16 & transposed, b128 load + b128 store
            const int n = t >> 2, kg = (t & 3) * 8;
            *(bf16x8*)(&sBT[n][kg]) =
                *(const bf16x8*)(WT + (size_t)(col0 + n) * 256 + k0 + kg);
        }
        __syncthreads();
        const bf16x16 a = frag_a(&sA[tr * 16][0], 32);
        acc0 = wmma_bf16(a, frag_b(&sBT[tc0 * 16][0], 32), acc0);
        acc1 = wmma_bf16(a, frag_b(&sBT[tc1 * 16][0], 32), acc1);
        __syncthreads();
    }

    { // park C in LDS (disjoint from sA/sBT: no extra barrier needed first)
        const int r = lane & 15, hi = lane >> 4;
#pragma unroll
        for (int i = 0; i < 8; ++i) {
            sOut[tr * 16 + i + 8 * hi][tc0 * 16 + r] = acc0[i];
            sOut[tr * 16 + i + 8 * hi][tc1 * 16 + r] = acc1[i];
        }
    }
    __syncthreads();

    const int b = row0 >> 6;                     // whole block is one batch slice
    if (col0 < 512) {                            // ---- q or k: row-major copy
        const int h = (col0 >> 6) & 3;
        bf16* dst = (col0 < 256 ? q : k) + ((size_t)(b * 4 + h)) * 4096;
        const int n = t >> 2, dg = (t & 3) * 16;
        bf16x8 o0, o1;
#pragma unroll
        for (int j = 0; j < 8; ++j) {
            o0[j] = (bf16)(sOut[n][dg + j]     + sBias[dg + j]);
            o1[j] = (bf16)(sOut[n][dg + 8 + j] + sBias[dg + 8 + j]);
        }
        *(bf16x8*)(dst + (size_t)n * 64 + dg)     = o0;
        *(bf16x8*)(dst + (size_t)n * 64 + dg + 8) = o1;
    } else {                                     // ---- v: transpose via LDS
        const int h = (col0 - 512) >> 6;
        bf16* dst = vt + ((size_t)(b * 4 + h)) * 4096;
        const int d = t >> 2, ng = (t & 3) * 16;
        const float bb = sBias[d];
        bf16x8 o0, o1;
#pragma unroll
        for (int j = 0; j < 8; ++j) {
            o0[j] = (bf16)(sOut[ng + j][d]     + bb);
            o1[j] = (bf16)(sOut[ng + 8 + j][d] + bb);
        }
        *(bf16x8*)(dst + (size_t)d * 64 + ng)     = o0;
        *(bf16x8*)(dst + (size_t)d * 64 + ng + 8) = o1;
    }
}

// ---------------------------------------------------------------------------
// Kernel 2: rel[b,h,n,m] = mask[b,m] ? scale * dot(q[b,h,n,:], bias[b,n,m,h,:])
//                                    : -1e9
// One block per (b,n) => owns a contiguous 64KB slab of bias_features.
// Thread t <-> (m = t>>2, h = t&3): reads its own contiguous 256B run, so a
// wave streams contiguous 8KB — full HBM bandwidth on the 268MB stream.
// Grid 4096, block 256.  (fp32 VALU; 0.5 flop/byte, bandwidth-bound)
// ---------------------------------------------------------------------------
__global__ __launch_bounds__(256) void rel_kernel(
    const float* __restrict__ bias, const unsigned char* __restrict__ mask,
    const bf16* __restrict__ q, float* __restrict__ rel)
{
    __shared__ float sQf[4][64];
    const int bn = blockIdx.x;
    const int b = bn >> 6, n = bn & 63;
    const int t = threadIdx.x;
    { // stage q rows for all 4 heads (bf16 -> f32)
        const int h = t >> 6, d = t & 63;
        sQf[h][d] = (float)q[(((size_t)(b * 4 + h) * 64 + n) << 6) + d];
    }
    __syncthreads();

    const int m = t >> 2, h = t & 3;
    const float* p = bias + ((size_t)bn * 64 + m) * 256 + h * 64;
    float s = 0.f;
#pragma unroll
    for (int j = 0; j < 64; j += 4) {
        const float4 bv = *(const float4*)(p + j);
        s = fmaf(bv.x, sQf[h][j + 0], s);
        s = fmaf(bv.y, sQf[h][j + 1], s);
        s = fmaf(bv.z, sQf[h][j + 2], s);
        s = fmaf(bv.w, sQf[h][j + 3], s);
    }
    const float out = mask[b * 64 + m] ? s * 0.125f : -1e9f;
    rel[(((size_t)(b * 4 + h) * 64 + n) << 6) + m] = out;
}

// ---------------------------------------------------------------------------
// Kernel 3: per (b,h): S = scale*Q@K^T + rel; P = softmax(S); O = P@V.
// Grid 256 (= b*4+h), block 256 (8 waves x 2 WMMA tiles). All tiles in LDS.
// ---------------------------------------------------------------------------
__global__ __launch_bounds__(256) void attn_kernel(
    const float* __restrict__ rel,
    const bf16* __restrict__ q, const bf16* __restrict__ k,
    const bf16* __restrict__ vt, bf16* __restrict__ aout)
{
    __shared__ __align__(16) bf16 sQ[64][64];    // [n][d]
    __shared__ __align__(16) bf16 sK[64][64];    // [m][d]  (== B^T tile for Q@K^T)
    __shared__ __align__(16) bf16 sVT[64][64];   // [d][m]  (== B^T tile for P@V)
    __shared__ __align__(16) bf16 sP[64][64];    // softmax probs, bf16
    __shared__ __align__(16) float sS[64][64];   // scores fp32; reused for O
    const int bh = blockIdx.x;
    const int b = bh >> 2, h = bh & 3;
    const size_t base = (size_t)bh * 4096;
    const int t = threadIdx.x, lane = t & 31, wave = t >> 5;

    { // stage q/k/v^T tiles (16 bf16 per thread per array)
        const int e = t * 16;
        bf16* fQ = &sQ[0][0]; bf16* fK = &sK[0][0]; bf16* fV = &sVT[0][0];
        *(bf16x8*)(fQ + e)     = *(const bf16x8*)(q  + base + e);
        *(bf16x8*)(fQ + e + 8) = *(const bf16x8*)(q  + base + e + 8);
        *(bf16x8*)(fK + e)     = *(const bf16x8*)(k  + base + e);
        *(bf16x8*)(fK + e + 8) = *(const bf16x8*)(k  + base + e + 8);
        *(bf16x8*)(fV + e)     = *(const bf16x8*)(vt + base + e);
        *(bf16x8*)(fV + e + 8) = *(const bf16x8*)(vt + base + e + 8);
    }
    __syncthreads();

    const int idx0 = wave * 2, idx1 = idx0 + 1;
    const int tr = idx0 >> 2;                    // same row-tile for the pair
    const int tc0 = idx0 & 3, tc1 = idx1 & 3;
    const int r = lane & 15, hi = lane >> 4;

    // ---- scores: S = 0.125 * Q @ K^T + rel -------------------------------
    v8f s0 = {}, s1 = {};
#pragma unroll
    for (int k0 = 0; k0 < 64; k0 += 32) {
        const bf16x16 a = frag_a(&sQ[tr * 16][k0], 64);
        s0 = wmma_bf16(a, frag_b(&sK[tc0 * 16][k0], 64), s0);
        s1 = wmma_bf16(a, frag_b(&sK[tc1 * 16][k0], 64), s1);
    }
    const float* relb = rel + base;
#pragma unroll
    for (int i = 0; i < 8; ++i) {
        const int n = tr * 16 + i + 8 * hi;
        const int m0 = tc0 * 16 + r, m1 = tc1 * 16 + r;
        sS[n][m0] = s0[i] * 0.125f + relb[n * 64 + m0];
        sS[n][m1] = s1[i] * 0.125f + relb[n * 64 + m1];
    }
    __syncthreads();

    // ---- softmax over m (64 values/row): one wave handles 8 rows ---------
#pragma unroll
    for (int rr = 0; rr < 8; ++rr) {
        const int n = wave * 8 + rr;
        const float e0 = sS[n][lane], e1 = sS[n][lane + 32];
        float mx = fmaxf(e0, e1);
#pragma unroll
        for (int off = 16; off; off >>= 1) mx = fmaxf(mx, __shfl_xor(mx, off, 32));
        const float p0 = __expf(e0 - mx), p1 = __expf(e1 - mx);
        float sm = p0 + p1;
#pragma unroll
        for (int off = 16; off; off >>= 1) sm += __shfl_xor(sm, off, 32);
        const float inv = 1.0f / sm;
        sP[n][lane]      = (bf16)(p0 * inv);
        sP[n][lane + 32] = (bf16)(p1 * inv);
    }
    __syncthreads();

    // ---- O = P @ V  (B^T tile is sVT[d][m], K=m contiguous) ---------------
    v8f o0 = {}, o1 = {};
#pragma unroll
    for (int k0 = 0; k0 < 64; k0 += 32) {
        const bf16x16 a = frag_a(&sP[tr * 16][k0], 64);
        o0 = wmma_bf16(a, frag_b(&sVT[tc0 * 16][k0], 64), o0);
        o1 = wmma_bf16(a, frag_b(&sVT[tc1 * 16][k0], 64), o1);
    }
    // park O in sS (everyone is past their softmax reads), then contiguous out
#pragma unroll
    for (int i = 0; i < 8; ++i) {
        const int n = tr * 16 + i + 8 * hi;
        sS[n][tc0 * 16 + r] = o0[i];
        sS[n][tc1 * 16 + r] = o1[i];
    }
    __syncthreads();
    {
        const int n = t >> 2, dg = (t & 3) * 16;
        bf16x8 a0, a1;
#pragma unroll
        for (int j = 0; j < 8; ++j) {
            a0[j] = (bf16)sS[n][dg + j];
            a1[j] = (bf16)sS[n][dg + 8 + j];
        }
        bf16* dst = aout + ((size_t)(b * 64 + n)) * 256 + h * 64 + dg;
        *(bf16x8*)(dst)     = a0;
        *(bf16x8*)(dst + 8) = a1;
    }
}

// ---------------------------------------------------------------------------
// Kernel 4: out(4096x256 f32) = aout(bf16) @ Wproj + bproj
// Grid (4, 64). Same WMMA tiling; epilogue round-trips LDS -> float4 stores.
// ---------------------------------------------------------------------------
__global__ __launch_bounds__(256) void proj_gemm(
    const bf16* __restrict__ a, const bf16* __restrict__ WT,
    const float* __restrict__ bproj, float* __restrict__ out)
{
    __shared__ __align__(16) bf16  sA[64][32];
    __shared__ __align__(16) bf16  sBT[64][32];
    __shared__ __align__(16) float sOut[64][64];
    __shared__ float sBias[64];
    const int t = threadIdx.x;
    const int wave = t >> 5, lane = t & 31;
    const int row0 = blockIdx.y * 64;
    const int col0 = blockIdx.x * 64;
    const int idx0 = wave * 2, idx1 = idx0 + 1;
    const int tr = idx0 >> 2;
    const int tc0 = idx0 & 3, tc1 = idx1 & 3;
    v8f acc0 = {}, acc1 = {};

    if (t < 64) sBias[t] = bproj[col0 + t];

    for (int k0 = 0; k0 < 256; k0 += 32) {
        { // A already bf16
            const int m = t >> 2, kg = (t & 3) * 8;
            *(bf16x8*)(&sA[m][kg]) =
                *(const bf16x8*)(a + (size_t)(row0 + m) * 256 + k0 + kg);
        }
        { // B^T already bf16 & transposed
            const int n = t >> 2, kg = (t & 3) * 8;
            *(bf16x8*)(&sBT[n][kg]) =
                *(const bf16x8*)(WT + (size_t)(col0 + n) * 256 + k0 + kg);
        }
        __syncthreads();
        const bf16x16 af = frag_a(&sA[tr * 16][0], 32);
        acc0 = wmma_bf16(af, frag_b(&sBT[tc0 * 16][0], 32), acc0);
        acc1 = wmma_bf16(af, frag_b(&sBT[tc1 * 16][0], 32), acc1);
        __syncthreads();
    }

    {
        const int r = lane & 15, hi = lane >> 4;
#pragma unroll
        for (int i = 0; i < 8; ++i) {
            sOut[tr * 16 + i + 8 * hi][tc0 * 16 + r] = acc0[i];
            sOut[tr * 16 + i + 8 * hi][tc1 * 16 + r] = acc1[i];
        }
    }
    __syncthreads();
    {
        const int n = t >> 2, dg = (t & 3) * 16;
        float* dst = out + (size_t)(row0 + n) * 256 + col0 + dg;
#pragma unroll
        for (int v = 0; v < 4; ++v) {
            float4 o;
            o.x = sOut[n][dg + v * 4 + 0] + sBias[dg + v * 4 + 0];
            o.y = sOut[n][dg + v * 4 + 1] + sBias[dg + v * 4 + 1];
            o.z = sOut[n][dg + v * 4 + 2] + sBias[dg + v * 4 + 2];
            o.w = sOut[n][dg + v * 4 + 3] + sBias[dg + v * 4 + 3];
            *(float4*)(dst + v * 4) = o;
        }
    }
}

// ---------------------------------------------------------------------------
// Host launch: B=4,T=16,N=64,D=256 => 4096 token rows.
// Workspace: q/k/v^T/aout bf16 (2MB each) + rel f32 (4MB) + W^T bf16 (0.5MB).
// ---------------------------------------------------------------------------
extern "C" void kernel_launch(void* const* d_in, const int* in_sizes, int n_in,
                              void* d_out, int out_size, void* d_ws, size_t ws_size,
                              hipStream_t stream) {
    (void)in_sizes; (void)n_in; (void)out_size; (void)ws_size;
    const float*         x     = (const float*)d_in[0];
    const float*         biasf = (const float*)d_in[1];
    const unsigned char* mask  = (const unsigned char*)d_in[2]; // jnp bool_ = 1 byte
    const float*         Wqkv  = (const float*)d_in[3];
    const float*         bqkv  = (const float*)d_in[4];
    const float*         Wproj = (const float*)d_in[5];
    const float*         bproj = (const float*)d_in[6];
    float*               out   = (float*)d_out;

    bf16*  qws    = (bf16*)d_ws;                 // 1M elems each
    bf16*  kws    = qws  + (1u << 20);
    bf16*  vtws   = kws  + (1u << 20);
    bf16*  aows   = vtws + (1u << 20);
    float* relws  = (float*)(aows + (1u << 20)); // 1M f32
    bf16*  wqkvT  = (bf16*)(relws + (1u << 20)); // 768*256
    bf16*  wprojT = wqkvT + 768 * 256;           // 256*256

    transpose_weights<<<dim3(1024), 256, 0, stream>>>(Wqkv, Wproj, wqkvT, wprojT);
    qkv_gemm  <<<dim3(12, 64), 256, 0, stream>>>(x, wqkvT, bqkv, qws, kws, vtws);
    rel_kernel<<<dim3(4096),   256, 0, stream>>>(biasf, mask, qws, relws);
    attn_kernel<<<dim3(256),   256, 0, stream>>>(relws, qws, kws, vtws, aows);
    proj_gemm <<<dim3(4, 64),  256, 0, stream>>>(aows, wprojT, bproj, out);
}
